// Multi_Head_Attention_5677946765801
// MI455X (gfx1250) — matrix operand order repaired and checked by reading, not compiled, not run on hardware
//
#include <hip/hip_runtime.h>
#include <hip/hip_bf16.h>

// ---------------------------------------------------------------------------
// Shapes (fixed by the reference)
// ---------------------------------------------------------------------------
#define B_    8
#define S_    2048
#define DM_   1024
#define DK_   512
#define DV_   512
#define DOUT_ 1024

typedef __attribute__((ext_vector_type(16))) _Float16 v16h;
typedef __attribute__((ext_vector_type(8)))  _Float16 v8h;
typedef __attribute__((ext_vector_type(4)))  _Float16 v4h;
typedef __attribute__((ext_vector_type(8)))  float    v8f;

union V16 { v16h v; v8h h[2]; };

// ---------------------------------------------------------------------------
// Generic 128x128x32 WMMA GEMM.  256 threads = 8 waves (wave32), each wave
// computes a 32x64 tile as 2x4 v_wmma_f32_16x16x32_f16 accumulators.
//
//   A_F32  : A operand is f32 in global (converted to f16 while staging to LDS)
//   BLAYOUT: 0 = B is (K,N) f32 row-major   (weight matrices)
//            1 = B is (K,N) f16 row-major   (vn for attn @ V)
//            2 = B is (N,K) f16 row-major   (kn for Q @ K^T)
//   EPI    : 0 = f32 store, 1 = f32 + bias, 2 = f16 store, 3 = f16 + bias + relu
// ---------------------------------------------------------------------------
template<bool A_F32, int BLAYOUT, int EPI>
__global__ __launch_bounds__(256) void gemm128_wmma(
    const void* __restrict__ Ap, const void* __restrict__ Bp,
    const float* __restrict__ bias, void* __restrict__ Cp,
    int K, int lda, int ldb, int ldc,
    long long sA, long long sB, long long sC)
{
    constexpr int BM = 128, BN = 128, BK = 32, LST = 40;  // 40-half row stride: conflict-free
    __shared__ __align__(16) _Float16 As[BM * LST];
    __shared__ __align__(16) _Float16 Bs[BN * LST];

    const int tid  = threadIdx.x;
    const int lane = tid & 31;
    const int wave = tid >> 5;
    const int wm   = wave & 3;     // 4 wave rows  (32 each)
    const int wn   = wave >> 2;    // 2 wave cols  (64 each)
    const long long zb = blockIdx.z;
    const int m0 = blockIdx.y * BM;
    const int n0 = blockIdx.x * BN;

    const float*    A32 = (const float*)Ap    + zb * sA;
    const _Float16* A16 = (const _Float16*)Ap + zb * sA;
    const float*    B32 = (const float*)Bp    + zb * sB;
    const _Float16* B16 = (const _Float16*)Bp + zb * sB;

    v8f acc[2][4] = {};

    for (int k0 = 0; k0 < K; k0 += BK) {
        __syncthreads();
        // ---- stage A tile (BM x BK) -> As[m][k] (f16) ----
        if constexpr (A_F32) {
            #pragma unroll
            for (int p = 0; p < 4; ++p) {
                int s   = tid + p * 256;        // 1024 float4 slots
                int row = s >> 3;
                int c   = (s & 7) << 2;
                const float4 f = *(const float4*)(A32 + (long long)(m0 + row) * lda + k0 + c);
                v4h h; h[0] = (_Float16)f.x; h[1] = (_Float16)f.y;
                       h[2] = (_Float16)f.z; h[3] = (_Float16)f.w;
                *(v4h*)&As[row * LST + c] = h;
            }
        } else {
            #pragma unroll
            for (int p = 0; p < 2; ++p) {
                int s   = tid * 2 + p;          // 512 v8h slots
                int row = s >> 2;
                int c   = (s & 3) << 3;
                *(v8h*)&As[row * LST + c] =
                    *(const v8h*)(A16 + (long long)(m0 + row) * lda + k0 + c);
            }
        }
        // ---- stage B tile -> Bs[n][k] (f16, transposed for contiguous frag reads) ----
        if constexpr (BLAYOUT == 0) {           // (K,N) f32
            #pragma unroll
            for (int p = 0; p < 4; ++p) {
                int s  = tid + p * 256;         // 1024 float4 slots: 32 k-rows x 32
                int kk = s >> 5;
                int c  = (s & 31) << 2;
                const float4 f = *(const float4*)(B32 + (long long)(k0 + kk) * ldb + n0 + c);
                Bs[(c + 0) * LST + kk] = (_Float16)f.x;
                Bs[(c + 1) * LST + kk] = (_Float16)f.y;
                Bs[(c + 2) * LST + kk] = (_Float16)f.z;
                Bs[(c + 3) * LST + kk] = (_Float16)f.w;
            }
        } else if constexpr (BLAYOUT == 1) {    // (K,N) f16
            #pragma unroll
            for (int p = 0; p < 2; ++p) {
                int s  = tid * 2 + p;           // 512 v8h slots: 32 k-rows x 16
                int kk = s >> 4;
                int c  = (s & 15) << 3;
                v8h f = *(const v8h*)(B16 + (long long)(k0 + kk) * ldb + n0 + c);
                #pragma unroll
                for (int e = 0; e < 8; ++e) Bs[(c + e) * LST + kk] = f[e];
            }
        } else {                                // (N,K) f16  (Q @ K^T path)
            #pragma unroll
            for (int p = 0; p < 2; ++p) {
                int s   = tid * 2 + p;
                int row = s >> 2;
                int c   = (s & 3) << 3;
                *(v8h*)&Bs[row * LST + c] =
                    *(const v8h*)(B16 + (long long)(n0 + row) * ldb + k0 + c);
            }
        }
        __syncthreads();

        // ---- load fragments per ISA 16-bit lane layouts, then 8 WMMAs ----
        const int ka = (lane >> 4) << 3;        // A: half-wave K offset 0 / 8 (pairs), +16 block
        const int kb = (lane >> 4) << 3;        // B: half-wave K offset 0 / 8 (pairs), +16 block
        V16 a[2], b[4];
        #pragma unroll
        for (int mt = 0; mt < 2; ++mt) {
            const _Float16* r = &As[(wm * 32 + mt * 16 + (lane & 15)) * LST];
            a[mt].h[0] = *(const v8h*)(r + ka);
            a[mt].h[1] = *(const v8h*)(r + 16 + ka);
        }
        #pragma unroll
        for (int nt = 0; nt < 4; ++nt) {
            const _Float16* r = &Bs[(wn * 64 + nt * 16 + (lane & 15)) * LST];
            b[nt].h[0] = *(const v8h*)(r + kb);
            b[nt].h[1] = *(const v8h*)(r + 16 + kb);
        }
        #pragma unroll
        for (int mt = 0; mt < 2; ++mt)
            #pragma unroll
            for (int nt = 0; nt < 4; ++nt)
                acc[mt][nt] = __builtin_amdgcn_wmma_f32_16x16x32_f16(
                    false, a[mt].v, false, b[nt].v, (short)0, acc[mt][nt], false, false);
    }

    // ---- epilogue: C/D layout -> VGPR r holds row (r + 8*(lane>=16)), col = lane&15 ----
    #pragma unroll
    for (int mt = 0; mt < 2; ++mt) {
        #pragma unroll
        for (int nt = 0; nt < 4; ++nt) {
            const int col = n0 + wn * 64 + nt * 16 + (lane & 15);
            const int rb  = m0 + wm * 32 + mt * 16 + ((lane >> 4) << 3);
            float bv = 0.0f;
            if constexpr (EPI == 1 || EPI == 3) bv = bias[col];
            #pragma unroll
            for (int r = 0; r < 8; ++r) {
                float val = acc[mt][nt][r] + bv;
                if constexpr (EPI == 3) val = fmaxf(val, 0.0f);
                const long long idx = zb * sC + (long long)(rb + r) * ldc + col;
                if constexpr (EPI == 2 || EPI == 3) ((_Float16*)Cp)[idx] = (_Float16)val;
                else                                ((float*)Cp)[idx]    = val;
            }
        }
    }
}

// ---------------------------------------------------------------------------
// LayerNorm over D=512 (one block per row), fused affine + optional 1/sqrt(dk),
// emits f16 operand for the WMMA GEMMs.
// ---------------------------------------------------------------------------
__global__ __launch_bounds__(256) void layernorm512(
    const float* __restrict__ x, const float* __restrict__ g,
    const float* __restrict__ be, _Float16* __restrict__ y, float scale)
{
    const int tid = threadIdx.x;
    const long long base = (long long)blockIdx.x * 512;
    const float v0 = x[base + tid], v1 = x[base + tid + 256];
    __shared__ float rs[256], rq[256];
    rs[tid] = v0 + v1;
    rq[tid] = v0 * v0 + v1 * v1;
    __syncthreads();
    for (int off = 128; off > 0; off >>= 1) {
        if (tid < off) { rs[tid] += rs[tid + off]; rq[tid] += rq[tid + off]; }
        __syncthreads();
    }
    const float mean = rs[0] * (1.0f / 512.0f);
    const float var  = rq[0] * (1.0f / 512.0f) - mean * mean;
    const float rinv = rsqrtf(var + 1e-6f);
    y[base + tid]       = (_Float16)(((v0 - mean) * rinv * g[tid]       + be[tid])       * scale);
    y[base + tid + 256] = (_Float16)(((v1 - mean) * rinv * g[tid + 256] + be[tid + 256]) * scale);
}

// ---------------------------------------------------------------------------
// Fused mask + softmax over 2048 scores (one block per query row).
// In-place on the f32 attn region of d_out, plus an f16 copy for attn @ V.
// ---------------------------------------------------------------------------
__global__ __launch_bounds__(256) void softmax2048(
    float* __restrict__ sc, const int* __restrict__ mask, _Float16* __restrict__ ah)
{
    const int tid = threadIdx.x;
    const long long base = (long long)blockIdx.x * 2048;
    float v[8];
    float mx = -3.0e38f;
    #pragma unroll
    for (int j = 0; j < 8; ++j) {
        const long long i = base + tid + j * 256;
        float s = sc[i];
        if (mask[i] == 0) s = -1.0e9f;
        v[j] = s;
        mx = fmaxf(mx, s);
    }
    __shared__ float red[256];
    red[tid] = mx; __syncthreads();
    for (int off = 128; off > 0; off >>= 1) {
        if (tid < off) red[tid] = fmaxf(red[tid], red[tid + off]);
        __syncthreads();
    }
    mx = red[0]; __syncthreads();
    float sum = 0.0f;
    #pragma unroll
    for (int j = 0; j < 8; ++j) { v[j] = __expf(v[j] - mx); sum += v[j]; }
    red[tid] = sum; __syncthreads();
    for (int off = 128; off > 0; off >>= 1) {
        if (tid < off) red[tid] += red[tid + off];
        __syncthreads();
    }
    const float inv = 1.0f / red[0];
    #pragma unroll
    for (int j = 0; j < 8; ++j) {
        const long long i = base + tid + j * 256;
        const float p = v[j] * inv;
        sc[i] = p;
        ah[i] = (_Float16)p;
    }
}

// ---------------------------------------------------------------------------
// Host-side orchestration (all on `stream`, graph-capture safe).
// ---------------------------------------------------------------------------
extern "C" void kernel_launch(void* const* d_in, const int* in_sizes, int n_in,
                              void* d_out, int out_size, void* d_ws, size_t ws_size,
                              hipStream_t stream)
{
    (void)in_sizes; (void)n_in; (void)out_size; (void)ws_size;

    const float* q    = (const float*)d_in[0];
    const float* k    = (const float*)d_in[1];
    const float* v    = (const float*)d_in[2];
    const int*   mask = (const int*)  d_in[3];
    const float* Wq   = (const float*)d_in[4];
    const float* bq   = (const float*)d_in[5];
    const float* Wk   = (const float*)d_in[6];
    const float* bk   = (const float*)d_in[7];
    const float* Wv   = (const float*)d_in[8];
    const float* bv   = (const float*)d_in[9];
    const float* g_q  = (const float*)d_in[10];
    const float* be_q = (const float*)d_in[11];
    const float* g_k  = (const float*)d_in[12];
    const float* be_k = (const float*)d_in[13];
    const float* g_v  = (const float*)d_in[14];
    const float* be_v = (const float*)d_in[15];
    const float* W1   = (const float*)d_in[16];
    const float* b1   = (const float*)d_in[17];
    const float* W2   = (const float*)d_in[18];
    const float* b2   = (const float*)d_in[19];

    // d_out = [ out (B,S,DOUT) | residual (B,S,H,DK) | attn (B,S,S) ], fp32
    float* outp     = (float*)d_out;
    float* residual = outp + (long long)B_ * S_ * DOUT_;
    float* attn32   = residual + (long long)B_ * S_ * DK_;

    // Workspace layout (bytes): ~208 MB total
    char* w = (char*)d_ws;
    float*    kp   = (float*)   (w);                   //  32 MB  kp f32
    float*    vp   = (float*)   (w +  33554432ll);     //  32 MB  vp f32
    _Float16* qn   = (_Float16*)(w +  67108864ll);     //  16 MB  LN(q)/temp f16
    _Float16* kn   = (_Float16*)(w +  83886080ll);     //  16 MB  LN(k) f16
    _Float16* vn   = (_Float16*)(w + 100663296ll);     //  16 MB  LN(v) f16
    _Float16* at16 = (_Float16*)(w + 117440512ll);     //  64 MB  attn f16
    _Float16* h0   = (_Float16*)(w + 184549376ll);     //  16 MB  attn@V f16
    _Float16* h1   = (_Float16*)(w + 201326592ll);     //  16 MB  relu(mlp1) f16

    const dim3 blk(256);
    const long long zero = 0;

    // 1) QKV projections: (16384 x 1024) @ (1024 x 512) + bias, fp32 out.
    //    qp goes directly into the residual slot of d_out.
    gemm128_wmma<true, 0, 1><<<dim3(DK_/128, (B_*S_)/128, 1), blk, 0, stream>>>(
        q, Wq, bq, residual, DM_, DM_, DK_, DK_, zero, zero, zero);
    gemm128_wmma<true, 0, 1><<<dim3(DK_/128, (B_*S_)/128, 1), blk, 0, stream>>>(
        k, Wk, bk, kp,       DM_, DM_, DK_, DK_, zero, zero, zero);
    gemm128_wmma<true, 0, 1><<<dim3(DV_/128, (B_*S_)/128, 1), blk, 0, stream>>>(
        v, Wv, bv, vp,       DM_, DM_, DV_, DV_, zero, zero, zero);

    // 2) Per-token LayerNorms; fold 1/sqrt(dk) into qn.
    const float inv_temp = 1.0f / sqrtf((float)DK_);
    layernorm512<<<B_*S_, 256, 0, stream>>>(residual, g_q, be_q, qn, inv_temp);
    layernorm512<<<B_*S_, 256, 0, stream>>>(kp,       g_k, be_k, kn, 1.0f);
    layernorm512<<<B_*S_, 256, 0, stream>>>(vp,       g_v, be_v, vn, 1.0f);

    // 3) scores = qn @ kn^T  (batched over B; NT GEMM), fp32 into d_out attn slot.
    gemm128_wmma<false, 2, 0><<<dim3(S_/128, S_/128, B_), blk, 0, stream>>>(
        qn, kn, nullptr, attn32, DK_, DK_, DK_, S_,
        (long long)S_ * DK_, (long long)S_ * DK_, (long long)S_ * S_);

    // 4) mask + softmax, in-place f32 + f16 copy for the next GEMM.
    softmax2048<<<B_*S_, 256, 0, stream>>>(attn32, mask, at16);

    // 5) attn @ vn  (batched NN GEMM, f16 A and B, f16 out).
    gemm128_wmma<false, 1, 2><<<dim3(DV_/128, S_/128, B_), blk, 0, stream>>>(
        at16, vn, nullptr, h0, S_, S_, DV_, DV_,
        (long long)S_ * S_, (long long)S_ * DV_, (long long)S_ * DV_);

    // 6) MLP: relu(h0 @ W1 + b1) @ W2 + b2 -> d_out main slot (fp32).
    gemm128_wmma<false, 0, 3><<<dim3(DV_/128, (B_*S_)/128, 1), blk, 0, stream>>>(
        h0, W1, b1, h1, DV_, DV_, DV_, DV_, zero, zero, zero);
    gemm128_wmma<false, 0, 1><<<dim3(DOUT_/128, (B_*S_)/128, 1), blk, 0, stream>>>(
        h1, W2, b2, outp, DV_, DV_, DOUT_, DOUT_, zero, zero, zero);
}
